// rnn_tangent_iso_78503412236566
// MI455X (gfx1250) — matrix operand order repaired and checked by reading, not compiled, not run on hardware
//
#include <hip/hip_runtime.h>
#include <hip/hip_bf16.h>

// Problem constants (match reference)
#define BB   1024
#define SS   256
#define HH   128
#define NSYM 6

typedef __bf16 bf16;
typedef __attribute__((ext_vector_type(16))) __bf16 v16bf;
typedef __attribute__((ext_vector_type(8)))  __bf16 v8bf;
typedef __attribute__((ext_vector_type(8)))  float  v8f;

// LDS strides chosen for bank-conflict-free 16-row access
#define XS 136   // x activation row stride in bf16 elems (272 B, 16B aligned, 4-bank lane step)
#define AS 40    // A_cat row stride in bf16 elems (80 B)
#define SB 8     // strain/stress row stride in f32 elems

// Load one 16x32 bf16 A-matrix fragment from LDS in this kernel's contiguous k order:
// lane L (row = L%16): VGPRs 0-7 hold K = k0 + (L/16)*16 .. +15 (contiguous, matches B)
__device__ __forceinline__ v16bf lds_afrag(const bf16* xb, int stride, int k0, int l16, int lhi) {
  const bf16* p = xb + l16 * stride + k0 + lhi * 16;
  v8bf lo = *(const v8bf*)p;          // ds_load_b128
  v8bf hi = *(const v8bf*)(p + 8);    // ds_load_b128
  return __builtin_shufflevector(lo, hi, 0, 1, 2, 3, 4, 5, 6, 7, 8, 9, 10, 11, 12, 13, 14, 15);
}

__global__ __launch_bounds__(256, 1)
void rnn_tangent_iso_kernel(const float* __restrict__ strain,   // (B,S,6)
                            const float* __restrict__ stress0,  // (B,1,6)
                            const float* __restrict__ w_in0,    // (128,6)
                            const float* __restrict__ b_in0,    // (128,)
                            const float* __restrict__ w_in1,    // (128,6)
                            const float* __restrict__ w_in2,    // (128,6)
                            const float* __restrict__ w_h,      // (2,128,128)
                            const float* __restrict__ b_h,      // (2,128)
                            const float* __restrict__ w_out,    // (2,128)
                            const float* __restrict__ b_out,    // (2,)
                            float* __restrict__ out)            // (B,S,6)
{
  __shared__ __align__(16) bf16  xbuf0[16 * XS];
  __shared__ __align__(16) bf16  xbuf1[16 * XS];
  // A_cat fixed slots: [0:6)=sbuf0 data, [6:12)=sbuf1 data, [12:18)=stress, rest zero
  __shared__ __align__(16) bf16  acat[16 * AS];
  __shared__ __align__(16) float sbuf[2][16 * SB];   // f32 strain ping-pong (for exact de)
  __shared__ __align__(16) float stressS[16 * SB];   // f32 running stress (the recurrence)
  __shared__ __align__(16) float klame[16 * 2];      // (lambda, mu-ish) per batch row

  const int tid   = threadIdx.x;
  const int wave  = tid >> 5;
  const int lane  = tid & 31;
  const int l16   = lane & 15;
  const int lhi   = lane >> 4;
  const int rbase = blockIdx.x * 16;   // 16 batch rows per workgroup

  // ---------------- preload weights as register-resident bf16 B fragments ----------------
  // B layout: lane = output column (n = wave*16 + l16), lanes 0-15 hold K=k0..k0+15,
  // lanes 16-31 hold K=k0+16..k0+31, two K values per VGPR.
  v16bf Bh[2][4];
  #pragma unroll
  for (int i = 0; i < 2; ++i) {
    #pragma unroll
    for (int f = 0; f < 4; ++f) {
      const float* src = w_h + ((size_t)(i * HH + wave * 16 + l16)) * HH + f * 32 + lhi * 16;
      v16bf r;
      #pragma unroll
      for (int e = 0; e < 16; ++e) r[e] = (bf16)src[e];
      Bh[i][f] = r;
    }
  }
  // Input layer as one K=32 fused matrix over A_cat's fixed slots.
  // Parity p: slot holding sbuf[p] gets w_in0 (prev strain), slot holding sbuf[p^1] gets w_in1.
  v16bf Bin0, Bin1;
  {
    const int n = wave * 16 + l16;
    #pragma unroll
    for (int e = 0; e < 16; ++e) {
      const int k = lhi * 16 + e;
      float v0 = 0.0f, v1 = 0.0f;
      if (k < 6)       { v0 = w_in0[n * NSYM + k];        v1 = w_in1[n * NSYM + k]; }
      else if (k < 12) { v0 = w_in1[n * NSYM + (k - 6)];  v1 = w_in0[n * NSYM + (k - 6)]; }
      else if (k < 18) { v0 = w_in2[n * NSYM + (k - 12)]; v1 = w_in2[n * NSYM + (k - 12)]; }
      Bin0[e] = (bf16)v0;
      Bin1[e] = (bf16)v1;
    }
  }
  // Output head (N padded 2 -> 16), only wave 7 uses it
  v16bf Bo[4];
  if (wave == 7) {
    #pragma unroll
    for (int f = 0; f < 4; ++f) {
      v16bf r;
      #pragma unroll
      for (int e = 0; e < 16; ++e) {
        const int k = f * 32 + lhi * 16 + e;
        r[e] = (bf16)((l16 < 2) ? w_out[l16 * HH + k] : 0.0f);
      }
      Bo[f] = r;
    }
  }
  const float biasIn  = b_in0[wave * 16 + l16];
  const float biasH0  = b_h[0 * HH + wave * 16 + l16];
  const float biasH1  = b_h[1 * HH + wave * 16 + l16];
  const float biasOut = (wave == 7 && l16 < 2) ? b_out[l16] : 0.0f;

  // ---------------- init: zero A_cat pad, sigma(t=0), first strain sample ----------------
  for (int e = tid; e < 16 * AS; e += 256) acat[e] = (bf16)0.0f;
  __syncthreads();
  if (tid < 96) {
    const int row = tid / 6, c = tid % 6;
    const int g = rbase + row;
    const float s0 = stress0[g * NSYM + c];
    stressS[row * SB + c] = s0;
    acat[row * AS + 12 + c] = (bf16)s0;
    out[((size_t)g * SS + 0) * NSYM + c] = s0;
    const float e0 = strain[((size_t)g * SS + 0) * NSYM + c];
    sbuf[0][row * SB + c] = e0;
    acat[row * AS + 0 + c] = (bf16)e0;   // slot 0 = sbuf[0] data (p=0 at t=0)
  }
  __syncthreads();

  #pragma unroll 2
  for (int t = 0; t < SS - 1; ++t) {
    const int p = t & 1;                       // folds to constant under unroll-2
    const float* epb = sbuf[p];
    float*       ecb = sbuf[p ^ 1];

    // stream in strain sample t+1 (f32 ping-pong + bf16 straight into A_cat); prefetch t+2
    if (tid < 96) {
      const int row = tid / 6, c = tid % 6;
      const int g = rbase + row;
      const float ecv = strain[((size_t)g * SS + (t + 1)) * NSYM + c];
      ecb[row * SB + c] = ecv;
      acat[row * AS + (p ^ 1) * 6 + c] = (bf16)ecv;
      if (t + 2 < SS)
        __builtin_prefetch(&strain[((size_t)g * SS + (t + 2)) * NSYM + c], 0, 0);
    }
    __syncthreads();

    // ---- input layer: x0 = relu(A_cat @ Wcat^T + b_in0), one K=32 WMMA per wave ----
    {
      v8f c;
      #pragma unroll
      for (int r = 0; r < 8; ++r) c[r] = biasIn;
      v16bf a = lds_afrag(acat, AS, 0, l16, lhi);
      c = __builtin_amdgcn_wmma_f32_16x16x32_bf16(false, a, false, p ? Bin1 : Bin0,
                                                  (short)0, c, false, false);
      #pragma unroll
      for (int r = 0; r < 8; ++r)
        xbuf0[(r + 8 * lhi) * XS + wave * 16 + l16] = (bf16)fmaxf(c[r], 0.0f);
    }
    __syncthreads();

    // ---- hidden layer 0: xbuf0 -> xbuf1 (4 WMMAs, K=128) ----
    {
      v8f c;
      #pragma unroll
      for (int r = 0; r < 8; ++r) c[r] = biasH0;
      #pragma unroll
      for (int f = 0; f < 4; ++f) {
        v16bf a = lds_afrag(xbuf0, XS, f * 32, l16, lhi);
        c = __builtin_amdgcn_wmma_f32_16x16x32_bf16(false, a, false, Bh[0][f], (short)0, c, false, false);
      }
      #pragma unroll
      for (int r = 0; r < 8; ++r)
        xbuf1[(r + 8 * lhi) * XS + wave * 16 + l16] = (bf16)fmaxf(c[r], 0.0f);
    }
    __syncthreads();

    // ---- hidden layer 1: xbuf1 -> xbuf0 ----
    {
      v8f c;
      #pragma unroll
      for (int r = 0; r < 8; ++r) c[r] = biasH1;
      #pragma unroll
      for (int f = 0; f < 4; ++f) {
        v16bf a = lds_afrag(xbuf1, XS, f * 32, l16, lhi);
        c = __builtin_amdgcn_wmma_f32_16x16x32_bf16(false, a, false, Bh[1][f], (short)0, c, false, false);
      }
      #pragma unroll
      for (int r = 0; r < 8; ++r)
        xbuf0[(r + 8 * lhi) * XS + wave * 16 + l16] = (bf16)fmaxf(c[r], 0.0f);
    }
    __syncthreads();

    // ---- output head (wave 7): klame = x @ w_out^T + b_out ----
    if (wave == 7) {
      v8f c;
      #pragma unroll
      for (int r = 0; r < 8; ++r) c[r] = biasOut;
      #pragma unroll
      for (int f = 0; f < 4; ++f) {
        v16bf a = lds_afrag(xbuf0, XS, f * 32, l16, lhi);
        c = __builtin_amdgcn_wmma_f32_16x16x32_bf16(false, a, false, Bo[f], (short)0, c, false, false);
      }
      if (l16 < 2) {
        #pragma unroll
        for (int r = 0; r < 8; ++r)
          klame[(r + 8 * lhi) * 2 + l16] = c[r];
      }
    }
    __syncthreads();

    // ---- stress update in f32: sigma += lambda*tr(de)*1_{j<3} + 2mu*de ----
    // Writes the f32 recurrence AND the bf16 copy into A_cat's stress slot.
    if (tid < 96) {
      const int row = tid / 6, j = tid % 6;
      const int g = rbase + row;
      const float lam   = klame[row * 2 + 0];
      const float twomu = 2.0f * klame[row * 2 + 1];
      const float de = ecb[row * SB + j] - epb[row * SB + j];
      float tr = 0.0f;
      #pragma unroll
      for (int c2 = 0; c2 < 3; ++c2) tr += ecb[row * SB + c2] - epb[row * SB + c2];
      float s = stressS[row * SB + j] + ((j < 3) ? lam * tr : 0.0f) + twomu * de;
      stressS[row * SB + j] = s;
      acat[row * AS + 12 + j] = (bf16)s;
      out[((size_t)g * SS + (t + 1)) * NSYM + j] = s;
    }
    __syncthreads();
  }
}

extern "C" void kernel_launch(void* const* d_in, const int* in_sizes, int n_in,
                              void* d_out, int out_size, void* d_ws, size_t ws_size,
                              hipStream_t stream) {
  (void)in_sizes; (void)n_in; (void)out_size; (void)d_ws; (void)ws_size;
  const float* strain  = (const float*)d_in[0];
  const float* stress0 = (const float*)d_in[1];
  const float* w_in0   = (const float*)d_in[2];
  const float* b_in0   = (const float*)d_in[3];
  const float* w_in1   = (const float*)d_in[4];
  const float* w_in2   = (const float*)d_in[5];
  const float* w_h     = (const float*)d_in[6];
  const float* b_h     = (const float*)d_in[7];
  const float* w_out   = (const float*)d_in[8];
  const float* b_out   = (const float*)d_in[9];
  float* out = (float*)d_out;

  dim3 grid(BB / 16);   // 64 workgroups, 16 batch rows each
  dim3 block(256);      // 8 wave32s: one 16-col slice of H per wave
  hipLaunchKernelGGL(rnn_tangent_iso_kernel, grid, block, 0, stream,
                     strain, stress0, w_in0, b_in0, w_in1, w_in2,
                     w_h, b_h, w_out, b_out, out);
}
